// MockLoHAAdapter_79465484911151
// MI455X (gfx1250) — matrix operand order repaired and checked by reading, not compiled, not run on hardware
//
#include <hip/hip_runtime.h>
#include <stdint.h>

// Problem constants (from the reference)
#define IN_DIM  4096
#define OUT_DIM 4096
#define M_TOT   8192      // BATCH*SEQ = 4*2048
#define RANK    16

// GEMM tiling
#define BLK_M 128
#define BLK_N 128
#define BLK_K 32
#define NK    (IN_DIM / BLK_K)
#define LDSP  40                 // padded LDS row stride in bf16 elems (80B, 16B-aligned)
#define TILE  (BLK_M * LDSP)     // ushorts per plane (5120)

typedef __attribute__((ext_vector_type(16))) __bf16       v16bf;
typedef __attribute__((ext_vector_type(8)))  float        v8f;
typedef __attribute__((ext_vector_type(4)))  unsigned int u32x4;
typedef __attribute__((ext_vector_type(4)))  float        f32x4;

union Frag  { v16bf v; u32x4 q[2]; };           // 32B: one WMMA 16-bit operand per lane
union Pack8 { unsigned short s[8]; u32x4 q; };  // 8 bf16 <-> 16B

__device__ __forceinline__ unsigned short f2bf(float f) {
    __bf16 h = (__bf16)f;                        // RNE convert
    return __builtin_bit_cast(unsigned short, h);
}
__device__ __forceinline__ float bf2f(unsigned short u) {
    return (float)__builtin_bit_cast(__bf16, u);
}

// ---- CDNA5 async global->LDS (ASYNCcnt-tracked, no VGPR round trip) ----
// lds_off: LDS byte address (low 32 bits of the flat pointer, per ISA 10.2
// aperture mapping). gaddr: global byte address.
__device__ __forceinline__ void async_ld_b128(uint32_t lds_off, uint64_t gaddr) {
    asm volatile("global_load_async_to_lds_b128 %0, %1, off"
                 :: "v"(lds_off), "v"(gaddr) : "memory");
}
__device__ __forceinline__ void wait_async_le4() {
    asm volatile("s_wait_asynccnt 0x4" ::: "memory");
}
__device__ __forceinline__ void wait_async_0() {
    asm volatile("s_wait_asynccnt 0x0" ::: "memory");
}

// ---------------------------------------------------------------------------
// Kernel 1: W_eff = org_weight + (w1a@w1b)*(w2a@w2b)*scalar, split to bf16 hi/lo
// ---------------------------------------------------------------------------
__global__ __launch_bounds__(256) void loha_prep(
    const float* __restrict__ orgW,
    const float* __restrict__ w1a, const float* __restrict__ w1b,
    const float* __restrict__ w2a, const float* __restrict__ w2b,
    const float* __restrict__ scalarp,
    unsigned short* __restrict__ Whi, unsigned short* __restrict__ Wlo)
{
    const int o = blockIdx.x;
    const float sc = scalarp[0];
    float a1[RANK], a2[RANK];
#pragma unroll
    for (int r = 0; r < RANK; ++r) {
        a1[r] = w1a[o * RANK + r];
        a2[r] = w2a[o * RANK + r];
    }
    for (int i = threadIdx.x; i < IN_DIM; i += 256) {
        float s1 = 0.f, s2 = 0.f;
#pragma unroll
        for (int r = 0; r < RANK; ++r) {
            s1 = fmaf(a1[r], w1b[r * IN_DIM + i], s1);
            s2 = fmaf(a2[r], w2b[r * IN_DIM + i], s2);
        }
        const size_t idx = (size_t)o * IN_DIM + i;
        const float w = orgW[idx] + s1 * s2 * sc;
        const unsigned short h = f2bf(w);
        const unsigned short l = f2bf(w - bf2f(h));
        Whi[idx] = h;
        Wlo[idx] = l;
    }
}

// ---------------------------------------------------------------------------
// Kernel 2: out = x @ W_eff^T + bias, bf16x3 split-precision WMMA GEMM.
// 256 threads (8 waves), 128x128 tile, K-step 32, double-buffered LDS.
// W tiles staged with global_load_async_to_lds_b128; x tiles loaded an
// iteration early into registers, converted to bf16 hi/lo after the barrier.
// ---------------------------------------------------------------------------
__global__ __launch_bounds__(256) void loha_gemm(
    const float* __restrict__ x,
    const unsigned short* __restrict__ Whi,
    const unsigned short* __restrict__ Wlo,
    const float* __restrict__ bias,
    float* __restrict__ out)
{
    // [buf][plane][...]: planes 0=A_hi 1=A_lo 2=B_hi 3=B_lo
    __shared__ unsigned short smem[2 * 4 * TILE];   // 80 KB

    const int t    = threadIdx.x;
    const int wid  = t >> 5;
    const int lane = t & 31;
    const int lh   = lane >> 4;    // K-half selector within fragment
    const int l16  = lane & 15;

    const int waveM = (wid & 3) * 32;   // local M base (2 mtiles)
    const int waveN = (wid >> 2) * 64;  // local N base (4 ntiles)

    const int blockM = blockIdx.y * BLK_M;
    const int blockN = blockIdx.x * BLK_N;

    // staging: thread t covers row t/2, k-halves [0..15] or [16..31]
    const int srow = t >> 1;
    const int skh  = (t & 1) * 16;
    const int sidx = srow * LDSP + skh;          // ushort index within a plane

    const float*          xg  = x   + (size_t)(blockM + srow) * IN_DIM + skh;
    const unsigned short* whg = Whi + (size_t)(blockN + srow) * IN_DIM + skh;
    const unsigned short* wlg = Wlo + (size_t)(blockN + srow) * IN_DIM + skh;

    // LDS byte addresses of this thread's B-plane staging slots (buffer 0)
    const uint32_t lds_bhi   = (uint32_t)(uintptr_t)&smem[2 * TILE + sidx];
    const uint32_t lds_blo   = (uint32_t)(uintptr_t)&smem[3 * TILE + sidx];
    const uint32_t bufstride = 4 * TILE * 2;     // bytes per double-buffer slab

    v8f acc[2][4];
#pragma unroll
    for (int i = 0; i < 2; ++i)
#pragma unroll
        for (int j = 0; j < 4; ++j)
#pragma unroll
            for (int r = 0; r < 8; ++r)
                acc[i][j][r] = 0.0f;

    f32x4 xr[4];

    // issue async W stage for K-step ks into buffer `buf` (4 x b128 per thread)
    auto stageW = [&](int ks, int buf) {
        const uint64_t gh  = (uint64_t)(uintptr_t)(whg + (size_t)ks * BLK_K);
        const uint64_t gl  = (uint64_t)(uintptr_t)(wlg + (size_t)ks * BLK_K);
        const uint32_t off = (uint32_t)buf * bufstride;
        async_ld_b128(lds_bhi + off,      gh);
        async_ld_b128(lds_bhi + off + 16, gh + 16);
        async_ld_b128(lds_blo + off,      gl);
        async_ld_b128(lds_blo + off + 16, gl + 16);
    };
    auto loadX = [&](int ks) {
        const f32x4* xv = (const f32x4*)(xg + (size_t)ks * BLK_K);
        xr[0] = xv[0]; xr[1] = xv[1]; xr[2] = xv[2]; xr[3] = xv[3];
    };
    // f32 -> bf16 hi/lo split, store to A-planes of buffer `buf`
    auto storeX = [&](int buf) {
        float fv[16] = { xr[0].x, xr[0].y, xr[0].z, xr[0].w,
                         xr[1].x, xr[1].y, xr[1].z, xr[1].w,
                         xr[2].x, xr[2].y, xr[2].z, xr[2].w,
                         xr[3].x, xr[3].y, xr[3].z, xr[3].w };
        Pack8 ph0, ph1, pl0, pl1;
#pragma unroll
        for (int j = 0; j < 16; ++j) {
            const float f = fv[j];
            const unsigned short h = f2bf(f);
            const unsigned short l = f2bf(f - bf2f(h));
            if (j < 8) { ph0.s[j]     = h; pl0.s[j]     = l; }
            else       { ph1.s[j - 8] = h; pl1.s[j - 8] = l; }
        }
        unsigned short* base = &smem[(size_t)buf * 4 * TILE];
        u32x4* dh = (u32x4*)&base[0 * TILE + sidx]; dh[0] = ph0.q; dh[1] = ph1.q;
        u32x4* dl = (u32x4*)&base[1 * TILE + sidx]; dl[0] = pl0.q; dl[1] = pl1.q;
    };

    // prologue: stage K-step 0 into buffer 0
    loadX(0);
    stageW(0, 0);
    storeX(0);

    for (int ks = 0; ks < NK; ++ks) {
        const int cur = ks & 1;
        const int nxt = cur ^ 1;
        const bool have_next = (ks + 1 < NK);

        if (have_next) {
            loadX(ks + 1);        // global f32 loads held in regs across barrier
            stageW(ks + 1, nxt);  // async global->LDS, lands behind compute
            wait_async_le4();     // cur's 4 asyncs done (in-order), next's pending
        } else {
            wait_async_0();
        }
        __syncthreads();          // cur tile fully staged & safe to read

        if (have_next) storeX(nxt);   // convert x(ks+1) into next buffer

        const unsigned short* base  = &smem[(size_t)cur * 4 * TILE];
        const unsigned short* As_hi = base;
        const unsigned short* As_lo = base + TILE;
        const unsigned short* Bs_hi = base + 2 * TILE;
        const unsigned short* Bs_lo = base + 3 * TILE;

        // B fragments: lane = column n, 16 contiguous K per lane
        Frag bh[4], bl[4];
#pragma unroll
        for (int nt = 0; nt < 4; ++nt) {
            const int nr = waveN + nt * 16 + l16;
            const u32x4* pbh = (const u32x4*)&Bs_hi[nr * LDSP + lh * 16];
            bh[nt].q[0] = pbh[0]; bh[nt].q[1] = pbh[1];
            const u32x4* pbl = (const u32x4*)&Bs_lo[nr * LDSP + lh * 16];
            bl[nt].q[0] = pbl[0]; bl[nt].q[1] = pbl[1];
        }
        // A fragments + 3x bf16 WMMA per tile (bf16x3 fp32 emulation)
#pragma unroll
        for (int mt = 0; mt < 2; ++mt) {
            const int mr = waveM + mt * 16 + l16;
            Frag ah, al;
            const u32x4* pah = (const u32x4*)&As_hi[mr * LDSP];
            ah.q[0] = pah[2 * lh]; ah.q[1] = pah[2 * lh + 1];
            const u32x4* pal = (const u32x4*)&As_lo[mr * LDSP];
            al.q[0] = pal[2 * lh]; al.q[1] = pal[2 * lh + 1];
#pragma unroll
            for (int nt = 0; nt < 4; ++nt) {
                v8f c = acc[mt][nt];
                c = __builtin_amdgcn_wmma_f32_16x16x32_bf16(false, ah.v, false, bh[nt].v, (short)0, c, false, false);
                c = __builtin_amdgcn_wmma_f32_16x16x32_bf16(false, ah.v, false, bl[nt].v, (short)0, c, false, false);
                c = __builtin_amdgcn_wmma_f32_16x16x32_bf16(false, al.v, false, bh[nt].v, (short)0, c, false, false);
                acc[mt][nt] = c;
            }
        }
        __syncthreads();          // all reads of cur done before it's restaged
    }

    // epilogue: C/D layout (VGPR r -> M = r + 8*lh, N = lane%16), fused bias
#pragma unroll
    for (int mt = 0; mt < 2; ++mt) {
        const int gmBase = blockM + waveM + mt * 16 + lh * 8;
#pragma unroll
        for (int nt = 0; nt < 4; ++nt) {
            const int gn = blockN + waveN + nt * 16 + l16;
            const float b = bias[gn];
#pragma unroll
            for (int r = 0; r < 8; ++r)
                out[(size_t)(gmBase + r) * OUT_DIM + gn] = acc[mt][nt][r] + b;
        }
    }
}

// ---------------------------------------------------------------------------
extern "C" void kernel_launch(void* const* d_in, const int* in_sizes, int n_in,
                              void* d_out, int out_size, void* d_ws, size_t ws_size,
                              hipStream_t stream) {
    const float* x      = (const float*)d_in[0];
    const float* orgW   = (const float*)d_in[1];
    const float* bias   = (const float*)d_in[2];
    const float* w1a    = (const float*)d_in[3];
    const float* w1b    = (const float*)d_in[4];
    const float* w2a    = (const float*)d_in[5];
    const float* w2b    = (const float*)d_in[6];
    const float* scalar = (const float*)d_in[7];

    // workspace: W_eff hi/lo bf16 planes, 2 * 4096*4096 * 2B = 64 MB
    unsigned short* Whi = (unsigned short*)d_ws;
    unsigned short* Wlo = Whi + (size_t)OUT_DIM * IN_DIM;

    loha_prep<<<OUT_DIM, 256, 0, stream>>>(orgW, w1a, w1b, w2a, w2b, scalar, Whi, Wlo);

    dim3 grid(OUT_DIM / BLK_N, M_TOT / BLK_M);  // (32, 64)
    loha_gemm<<<grid, 256, 0, stream>>>(x, Whi, Wlo, bias, (float*)d_out);
}
